// RLOfThoughtsModule_8555574854198
// MI455X (gfx1250) — matrix operand order repaired and checked by reading, not compiled, not run on hardware
//
#include <hip/hip_runtime.h>
#include <hip/hip_bf16.h>
#include <math.h>

// Problem constants (match reference)
#define BB   8
#define SS   4096
#define HH   1024
#define NAV  128
#define EE   8
#define FF   2048
#define MROWS 32   // rows per workgroup tile

typedef __attribute__((ext_vector_type(16))) _Float16 v16h;
typedef __attribute__((ext_vector_type(8)))  float    v8f;

static __device__ __forceinline__ float gelu_exact(float x) {
    return 0.5f * x * (1.0f + erff(x * 0.70710678118654752440f));
}

static __device__ __forceinline__ v8f vzero8() {
    v8f z;
    #pragma unroll
    for (int i = 0; i < 8; ++i) z[i] = 0.0f;
    return z;
}

// ---- CDNA5 async global->LDS copy (ASYNCcnt path), inline asm per ISA §10 ----
// global_load_async_to_lds_b128 vdst(lds byte addr), vaddr(64b global), off
static __device__ __forceinline__ void async_g2l_b128(void* lds_ptr, const void* gptr) {
    const unsigned lds_off = (unsigned)(uintptr_t)lds_ptr;   // low 32b = LDS offset
    const unsigned long long ga = (unsigned long long)(uintptr_t)gptr;
    asm volatile("global_load_async_to_lds_b128 %0, %1, off"
                 :: "v"(lds_off), "v"(ga) : "memory");
}
static __device__ __forceinline__ void wait_async0() {
    asm volatile("s_wait_asynccnt 0x0" ::: "memory");
}
// wait for previous slab while next slab (8 async ops/wave) stays in flight
static __device__ __forceinline__ void wait_async8() {
    asm volatile("s_wait_asynccnt 0x8" ::: "memory");
}

// A-fragment (16x32 f16) for v_wmma_f32_16x16x32_f16 from row-major f16 LDS.
// lane L holds row (L&15); halfs 0..7 = K kb..kb+7, halfs 8..15 = K kb+16..kb+23,
// kb = k0 + (L>=16 ? 8 : 0).  rowStride must be a multiple of 8 halfs.
static __device__ __forceinline__ v16h load_afrag(const _Float16* base, int rowStride,
                                                  int r0, int k0) {
    const int lane = threadIdx.x & 31;
    const int r  = r0 + (lane & 15);
    const int kb = k0 + ((lane >> 4) << 3);
    const _Float16* p = base + (size_t)r * rowStride + kb;
    union { v16h v; uint4 q[2]; } u;
    u.q[0] = *(const uint4*)(p);        // K kb .. kb+7
    u.q[1] = *(const uint4*)(p + 16);   // K kb+16 .. kb+23
    return u.v;
}

// B-fragment (32x16 f16) from column-major f16 LDS staging [col][k] (stride sk).
// lane L holds col (L&15); halfs 0..7 = K kb..kb+7, halfs 8..15 = K kb+16..kb+23, kb = k0 + (L>=16 ? 8 : 0).
static __device__ __forceinline__ v16h load_bfrag(const _Float16* base, int sk,
                                                  int c0, int k0) {
    const int lane = threadIdx.x & 31;
    const int c  = c0 + (lane & 15);
    const int kb = k0 + ((lane >> 4) << 3);
    const _Float16* p = base + (size_t)c * sk + kb;
    union { v16h v; uint4 q[2]; } u;
    u.q[0] = *(const uint4*)(p);
    u.q[1] = *(const uint4*)(p + 16);
    return u.v;
}

static __device__ __forceinline__ v8f wmma16(v16h a, v16h b, v8f c) {
    return __builtin_amdgcn_wmma_f32_16x16x32_f16(false, a, false, b, (short)0, c,
                                                  false, false);
}

// ---------------------------------------------------------------------------
// Kernel 1: gating -> expert selection per batch
// ---------------------------------------------------------------------------
__global__ __launch_bounds__(NAV) void gate_kernel(const float* __restrict__ X,
                                                   const float* __restrict__ Wp1,
                                                   const float* __restrict__ bp1,
                                                   const float* __restrict__ Wp2,
                                                   const float* __restrict__ bp2,
                                                   int* __restrict__ sel) {
    const int b = blockIdx.x;
    const int j = threadIdx.x;               // 0..127
    __shared__ float t[NAV];
    __shared__ float logits[EE];

    const float* last = X + ((size_t)b * SS + (SS - 1)) * HH;
    float acc = bp1[j];
    for (int k = 0; k < HH; ++k) acc += last[k] * Wp1[(size_t)k * NAV + j];
    t[j] = gelu_exact(acc);
    __syncthreads();

    if (j < EE) {
        float l = bp2[j];
        for (int k = 0; k < NAV; ++k) l += t[k] * Wp2[(size_t)k * EE + j];
        logits[j] = l;
    }
    __syncthreads();

    if (j == 0) {
        int best = 0;
        float bv = logits[0];
        #pragma unroll
        for (int e = 1; e < EE; ++e)
            if (logits[e] > bv) { bv = logits[e]; best = e; }
        sel[b] = best;
    }
}

// ---------------------------------------------------------------------------
// Kernel 2: fused expert FFN + output projection + residual
// grid = (SS/MROWS, BB), block = 256 threads (8 wave32)
// ---------------------------------------------------------------------------
// LDS carving (halfs):
//   XB0 [32][136] @0       (4352)      phase B: Rp  [32][136] @0
//   XB1 [32][136] @4352    (4352)      phase B: WoB [128][136]@4352 (17408)
//   WB  [64][136] @8704    (8704)
//   Hh  [32][72]  @17408   (2304)
//   W2B [128][72] @19712   (9216)
//   raw f32 async double-buffer @28928: 2 x 8192 f32 = 64 KB
#define XB0_OFF 0
#define XB1_OFF 4352
#define WB_OFF  8704
#define HHL_OFF 17408
#define W2B_OFF 19712
#define RAW_OFF 28928
#define SMEM_HALFS (RAW_OFF + 32768)

__global__ __launch_bounds__(256) void moe_kernel(const float* __restrict__ X,
                                                  const float* __restrict__ W1,
                                                  const float* __restrict__ b1,
                                                  const float* __restrict__ W2,
                                                  const float* __restrict__ b2,
                                                  const float* __restrict__ Wo,
                                                  const float* __restrict__ bo,
                                                  const int* __restrict__ sel,
                                                  float* __restrict__ out) {
    __shared__ _Float16 smem[SMEM_HALFS];

    const int b    = blockIdx.y;
    const int row0 = blockIdx.x * MROWS;
    const int tid  = threadIdx.x;
    const int lane = tid & 31;
    const int wv   = tid >> 5;        // 0..7
    const int mi   = wv & 1;          // row-tile (0..1)
    const int nw   = wv >> 1;         // col group (0..3)
    const int hi   = lane >> 4;       // half-wave select
    const int ln   = lane & 15;

    const int e = sel[b];
    const float* W1e = W1 + (size_t)e * HH * FF;
    const float* b1e = b1 + (size_t)e * FF;
    const float* W2e = W2 + (size_t)e * FF * HH;
    const float* b2e = b2 + (size_t)e * HH;
    const float* Xb  = X + ((size_t)b * SS + row0) * HH;

    _Float16* WBp  = smem + WB_OFF;   // [64][136]
    _Float16* Hhp  = smem + HHL_OFF;  // [32][72]
    _Float16* W2Bp = smem + W2B_OFF;  // [128][72]
    float*    rawBase = (float*)(smem + RAW_OFF);   // 2 x 8192 f32

    // async issue helpers (8 ops/wave per 32 KB slab, 16 for the 64 KB Wo slab)
    auto issueW1 = [&](int fc, int i, int buf) {
        float* dst = rawBase + buf * 8192;          // [128][64]
        for (int g = tid; g < 2048; g += 256) {
            const int r = g >> 4, c4 = (g & 15) << 2;
            async_g2l_b128(&dst[r * 64 + c4],
                           W1e + (size_t)(i * 128 + r) * FF + fc + c4);
        }
    };
    auto issueW2 = [&](int fc, int p, int buf) {
        float* dst = rawBase + buf * 8192;          // [64][128]
        for (int g = tid; g < 2048; g += 256) {
            const int r = g >> 5, c4 = (g & 31) << 2;
            async_g2l_b128(&dst[r * 128 + c4],
                           W2e + (size_t)(fc + r) * HH + p * 128 + c4);
        }
    };
    auto issueWo = [&](int p3, int q) {
        for (int g = tid; g < 4096; g += 256) {     // [128][128] across both bufs
            const int r = g >> 5, c4 = (g & 31) << 2;
            async_g2l_b128(&rawBase[r * 128 + c4],
                           Wo + (size_t)(p3 * 128 + r) * HH + q * 128 + c4);
        }
    };

    v8f racc[16];                     // reasoned: rows mi*16..+15, 16 col tiles
    #pragma unroll
    for (int i = 0; i < 16; ++i) racc[i] = vzero8();

    // ---------------- phase A: h = gelu(X@W1e + b1); reasoned += h@W2e -----
    for (int fc = 0; fc < FF; fc += 64) {
        v8f hacc = vzero8();
        issueW1(fc, 0, 0);                          // pipeline prologue
        for (int i = 0; i < 8; ++i) {               // K slabs of 128 (GEMM1)
            _Float16* XBp = smem + ((i & 1) ? XB1_OFF : XB0_OFF);
            // direct-stage X slice [32][128] -> f16 (overlaps async flight)
            for (int idx = tid; idx < 32 * 128; idx += 256) {
                const int r = idx >> 7, k = idx & 127;
                XBp[r * 136 + k] = (_Float16)Xb[(size_t)r * HH + i * 128 + k];
            }
            if (i < 7) { issueW1(fc, i + 1, (i + 1) & 1); wait_async8(); }
            else       { wait_async0(); }
            __syncthreads();
            const float* rw = rawBase + (i & 1) * 8192;
            for (int idx = tid; idx < 128 * 64; idx += 256) {
                const int c = idx & 63, k = idx >> 6;
                WBp[c * 136 + k] = (_Float16)rw[k * 64 + c];
            }
            __syncthreads();
            #pragma unroll
            for (int ks = 0; ks < 4; ++ks) {
                v16h a  = load_afrag(XBp, 136, mi * 16, ks * 32);
                v16h bb = load_bfrag(WBp, 136, nw * 16, ks * 32);
                hacc = wmma16(a, bb, hacc);
            }
        }
        // bias + gelu -> f16 H chunk in LDS
        {
            const float bias = b1e[fc + nw * 16 + ln];
            #pragma unroll
            for (int j = 0; j < 8; ++j) {
                const int r = mi * 16 + (hi << 3) + j;
                Hhp[r * 72 + nw * 16 + ln] = (_Float16)gelu_exact(hacc[j] + bias);
            }
        }
        __syncthreads();
        // GEMM2: reasoned[32][1024] += H[32][64] @ W2e[fc..fc+63, :]
        issueW2(fc, 0, 0);                          // pipeline prologue
        for (int p = 0; p < 8; ++p) {
            if (p < 7) { issueW2(fc, p + 1, (p + 1) & 1); wait_async8(); }
            else       { wait_async0(); }
            __syncthreads();
            const float* rw = rawBase + (p & 1) * 8192;
            for (int idx = tid; idx < 64 * 128; idx += 256) {
                const int c = idx & 127, k = idx >> 7;
                W2Bp[c * 72 + k] = (_Float16)rw[k * 128 + c];
            }
            __syncthreads();
            #pragma unroll
            for (int t2 = 0; t2 < 2; ++t2) {
                const int ni = nw + t2 * 4;
                #pragma unroll
                for (int ks = 0; ks < 2; ++ks) {
                    v16h a  = load_afrag(Hhp, 72, mi * 16, ks * 32);
                    v16h bb = load_bfrag(W2Bp, 72, ni * 16, ks * 32);
                    racc[p * 2 + t2] = wmma16(a, bb, racc[p * 2 + t2]);
                }
            }
        }
    }

    // ---------------- phase B: out = reasoned @ Wo --------------------------
    _Float16* Rp   = smem + XB0_OFF;  // [32][136]
    _Float16* WoBp = smem + XB1_OFF;  // [128][136]

    v8f oacc[16];
    #pragma unroll
    for (int i = 0; i < 16; ++i) oacc[i] = vzero8();

    for (int p3 = 0; p3 < 8; ++p3) {  // K panels of GEMM3 (reasoned cols)
        __syncthreads();
        #pragma unroll
        for (int t2 = 0; t2 < 2; ++t2) {
            const int cip = (nw + t2 * 4) * 16 + ln;
            const float bias = b2e[p3 * 128 + cip];
            #pragma unroll
            for (int j = 0; j < 8; ++j) {
                const int r = mi * 16 + (hi << 3) + j;
                Rp[r * 136 + cip] = (_Float16)(racc[p3 * 2 + t2][j] + bias);
            }
        }
        __syncthreads();
        issueWo(p3, 0);                             // pipeline prologue
        for (int q = 0; q < 8; ++q) {               // output column panels
            wait_async0();
            __syncthreads();
            for (int idx = tid; idx < 128 * 128; idx += 256) {
                const int c = idx & 127, k = idx >> 7;
                WoBp[c * 136 + k] = (_Float16)rawBase[k * 128 + c];
            }
            __syncthreads();
            if (q < 7) issueWo(p3, q + 1);          // overlap with WMMAs below
            #pragma unroll
            for (int t3 = 0; t3 < 2; ++t3) {
                const int ni = nw + t3 * 4;
                #pragma unroll
                for (int ks = 0; ks < 4; ++ks) {
                    v16h a  = load_afrag(Rp, 136, mi * 16, ks * 32);
                    v16h bb = load_bfrag(WoBp, 136, ni * 16, ks * 32);
                    oacc[q * 2 + t3] = wmma16(a, bb, oacc[q * 2 + t3]);
                }
            }
        }
    }

    // ---------------- epilogue: residual + bias + scale ---------------------
    float* outb = out + ((size_t)b * SS + row0) * HH;
    #pragma unroll
    for (int q = 0; q < 8; ++q) {
        #pragma unroll
        for (int t3 = 0; t3 < 2; ++t3) {
            const int c = q * 128 + (nw + t3 * 4) * 16 + ln;
            const float bov = bo[c];
            #pragma unroll
            for (int j = 0; j < 8; ++j) {
                const int r = mi * 16 + (hi << 3) + j;
                const float xv = Xb[(size_t)r * HH + c];
                outb[(size_t)r * HH + c] = xv + 0.3f * (oacc[q * 2 + t3][j] + bov);
            }
        }
    }
}

// ---------------------------------------------------------------------------
extern "C" void kernel_launch(void* const* d_in, const int* in_sizes, int n_in,
                              void* d_out, int out_size, void* d_ws, size_t ws_size,
                              hipStream_t stream) {
    const float* X   = (const float*)d_in[0];
    const float* Wp1 = (const float*)d_in[1];
    const float* bp1 = (const float*)d_in[2];
    const float* Wp2 = (const float*)d_in[3];
    const float* bp2 = (const float*)d_in[4];
    const float* W1  = (const float*)d_in[5];
    const float* b1  = (const float*)d_in[6];
    const float* W2  = (const float*)d_in[7];
    const float* b2  = (const float*)d_in[8];
    const float* Wo  = (const float*)d_in[9];
    const float* bo  = (const float*)d_in[10];
    float* out = (float*)d_out;
    int* sel = (int*)d_ws;

    gate_kernel<<<dim3(BB), dim3(NAV), 0, stream>>>(X, Wp1, bp1, Wp2, bp2, sel);
    moe_kernel<<<dim3(SS / MROWS, BB), dim3(256), 0, stream>>>(
        X, W1, b1, W2, b2, Wo, bo, sel, out);
}
